// RelPosMultiHeadAttention_37615323578439
// MI455X (gfx1250) — matrix operand order repaired and checked by reading, not compiled, not run on hardware
//
#include <hip/hip_runtime.h>
#include <hip/hip_bf16.h>

// ---------------- CDNA5 WMMA types & helpers (wave32) ----------------
typedef __bf16 bf16;
typedef __attribute__((ext_vector_type(16))) __bf16 v16bf;
typedef __attribute__((ext_vector_type(8)))  __bf16 v8bf;
typedef __attribute__((ext_vector_type(8)))  float  v8f;
typedef __attribute__((ext_vector_type(4)))  float  v4f;

#define H_  12
#define C_  768
#define OC_ 768
#define HD_ 64
#define B_  4
#define T_  1024
#define WIN_ 4
#define QSCALE 0.125f   // HD^-0.5

__device__ inline v8f wmma_bf(v16bf a, v16bf b, v8f c) {
  // D = A(16x32 bf16) * B(32x16 bf16) + C(16x16 f32)
  return __builtin_amdgcn_wmma_f32_16x16x32_bf16(false, a, false, b, (short)0, c,
                                                 false, false);
}

__device__ inline v16bf cat8(v8bf lo, v8bf hi) {
  return __builtin_shufflevector(lo, hi, 0, 1, 2, 3, 4, 5, 6, 7,
                                 8, 9, 10, 11, 12, 13, 14, 15);
}

// A fragment 16x32 bf16, src row-major element (m,k) at src[m*ld+k].
// Lanes 0-15 row M=lane, K={khi..khi+7, 16+khi..+7}, khi=8*(lane>=16).
// Two contiguous 16B runs per lane -> 2x ds_load_b128.  Requires ld%8==0.
__device__ inline v16bf load_a_bf16(const bf16* src, int ld) {
  const int lane = threadIdx.x & 31;
  const int m = lane & 15;
  const int khi = (lane >> 4) << 3;
  const bf16* r = src + m * ld + khi;
  return cat8(*(const v8bf*)r, *(const v8bf*)(r + 16));
}

// A fragment from fp32 source (softmax probabilities). 4x ds_load_b128 + cvt.
__device__ inline v16bf load_a_f32(const float* src, int ld) {
  const int lane = threadIdx.x & 31;
  const int m = lane & 15;
  const int khi = (lane >> 4) << 3;
  const float* r = src + m * ld + khi;
  const v4f a0 = *(const v4f*)r;
  const v4f a1 = *(const v4f*)(r + 4);
  const v4f b0 = *(const v4f*)(r + 16);
  const v4f b1 = *(const v4f*)(r + 20);
  v16bf o;
#pragma unroll
  for (int i = 0; i < 4; ++i) {
    o[i]      = (bf16)a0[i];
    o[4 + i]  = (bf16)a1[i];
    o[8 + i]  = (bf16)b0[i];
    o[12 + i] = (bf16)b1[i];
  }
  return o;
}

// B fragment 32x16 bf16 from K-major-per-lane source: element (k,n) at src[n*ld+k].
// Lane n=lane&15, K={khi..khi+7, 16+khi..+7}, khi=8*(lane>=16) -> 2x ds_load_b128.
__device__ inline v16bf load_bT_bf16(const bf16* src, int ld) {
  const int lane = threadIdx.x & 31;
  const int n = lane & 15;
  const int khi = (lane >> 4) << 3;
  const bf16* r = src + n * ld + khi;
  return cat8(*(const v8bf*)r, *(const v8bf*)(r + 16));
}

// ---------------- Kernel 1: fused QKV projection (1x1 conv) ----------------
// grid (T/128, C/128, B*3), block 256 (8 waves, 4x2), 128x128 tile,
// 8 WMMA per wave per K-slab.  Q,K -> (b,h,t,d) bf16 (Q pre-scaled); V -> (b,h,d,t).
__global__ __launch_bounds__(256) void qkv_proj_kernel(
    const float* __restrict__ x, const float* __restrict__ cc,
    const float* __restrict__ wq, const float* __restrict__ bq,
    const float* __restrict__ wk, const float* __restrict__ bk,
    const float* __restrict__ wv, const float* __restrict__ bv,
    bf16* __restrict__ qb, bf16* __restrict__ kb, bf16* __restrict__ vb) {
  const int t0 = blockIdx.x * 128;
  const int o0 = blockIdx.y * 128;
  const int which = blockIdx.z % 3;
  const int b = blockIdx.z / 3;

  const float* W; const float* bias; const float* src; bf16* out; float qs;
  if (which == 0)      { W = wq; bias = bq; src = x;  out = qb; qs = QSCALE; }
  else if (which == 1) { W = wk; bias = bk; src = cc; out = kb; qs = 1.0f;   }
  else                 { W = wv; bias = bv; src = cc; out = vb; qs = 1.0f;   }

  const int LD = 40;                  // padded, %8==0, 16B-aligned rows
  __shared__ bf16 Atile[128 * 40];    // [o][c]   128x32 used
  __shared__ bf16 Bt[128 * 40];       // [t][c]   128x32 used (transposed stage)

  const int tid = threadIdx.x;
  const int wave = tid >> 5;
  const int msub = (wave & 3) * 32;    // 4 m-subtiles of 32
  const int nsub = (wave >> 2) * 64;   // 2 n-subtiles of 64

  v8f acc[2][4] = {};

  for (int kk = 0; kk < C_; kk += 32) {
    // A tile: (o-row, 8-c chunk); contiguous global, b128 LDS store
#pragma unroll
    for (int e = tid; e < 512; e += 256) {
      const int r = e >> 2, ch = (e & 3) * 8;
      const float* gp = W + (size_t)(o0 + r) * C_ + kk + ch;
      v8bf w;
#pragma unroll
      for (int i = 0; i < 8; ++i) w[i] = (bf16)gp[i];
      *(v8bf*)(Atile + r * LD + ch) = w;
    }
    // B tile transposed: (t-col, 8-c chunk); lane-coalesced global, b128 store
#pragma unroll
    for (int e = tid; e < 512; e += 256) {
      const int t = e & 127, ch = (e >> 7) * 8;
      const float* gp = src + ((size_t)b * C_ + kk + ch) * T_ + t0 + t;
      v8bf w;
#pragma unroll
      for (int i = 0; i < 8; ++i) w[i] = (bf16)gp[(size_t)i * T_];
      *(v8bf*)(Bt + t * LD + ch) = w;
    }
    if (kk + 32 < C_) {
      __builtin_prefetch(W + (size_t)(o0 + (tid >> 2)) * C_ + kk + 32, 0, 1);
      __builtin_prefetch(src + ((size_t)b * C_ + kk + 32 + (tid >> 7) * 8) * T_ + t0 + (tid & 127), 0, 1);
    }
    __syncthreads();
    const v16bf a0 = load_a_bf16(Atile + msub * LD, LD);
    const v16bf a1 = load_a_bf16(Atile + (msub + 16) * LD, LD);
#pragma unroll
    for (int j = 0; j < 4; ++j) {
      const v16bf bj = load_bT_bf16(Bt + (nsub + j * 16) * LD, LD);
      acc[0][j] = wmma_bf(a0, bj, acc[0][j]);
      acc[1][j] = wmma_bf(a1, bj, acc[1][j]);
    }
    __syncthreads();
  }

  const int lane = tid & 31;
  const int mo = (lane >> 4) << 3;
  const int n = lane & 15;
#pragma unroll
  for (int i = 0; i < 2; ++i) {
#pragma unroll
    for (int j = 0; j < 4; ++j) {
#pragma unroll
      for (int r = 0; r < 8; ++r) {
        const int m = msub + i * 16 + r + mo;
        const int o = o0 + m;
        const int t = t0 + nsub + j * 16 + n;
        const float val = (acc[i][j][r] + bias[o]) * qs;
        const int h = o >> 6, d = o & 63;
        if (which == 2)
          out[(((size_t)b * H_ + h) * HD_ + d) * T_ + t] = (bf16)val;   // (b,h,d,t)
        else
          out[(((size_t)b * H_ + h) * T_ + t) * HD_ + d] = (bf16)val;   // (b,h,t,d)
      }
    }
  }
}

// ---------------- Kernel 2: relative-position attention ----------------
// grid (T/32, H, B), block 128 (4 waves). Dynamic LDS partitioned manually.
#define SLD_ 1032                                  // 1024 + 8 pad (f32)
#define SM_S    0                                  // 32*1032*4 = 132096
#define SM_QT   132096                             // 32*64*2   = 4096
#define SM_KVT  (132096 + 4096)                    // 128*64*2  = 16384
#define SM_RELK (SM_KVT + 16384)                   // 9*64*4    = 2304
#define SM_RELV (SM_RELK + 2304)                   // 2304
#define SM_OT   (SM_RELV + 2304)                   // 32*64*4   = 8192
#define SM_RED  (SM_OT + 8192)                     // 32*4*4    = 512
#define SM_TOTAL (SM_RED + 512)                    // 165888 bytes

__global__ __launch_bounds__(128) void attn_kernel(
    const bf16* __restrict__ qb, const bf16* __restrict__ kb,
    const bf16* __restrict__ vb, const float* __restrict__ mask,
    const float* __restrict__ erk, const float* __restrict__ erv,
    bf16* __restrict__ aout) {
  const int q0 = blockIdx.x * 32;
  const int h  = blockIdx.y;
  const int b  = blockIdx.z;
  const int tid = threadIdx.x;
  const int wave = tid >> 5;
  const int lane = tid & 31;

  extern __shared__ unsigned char smem[];
  float* S    = (float*)(smem + SM_S);     // [32][SLD_]
  bf16*  Qt   = (bf16*) (smem + SM_QT);    // [32][64]
  bf16*  KVt  = (bf16*) (smem + SM_KVT);   // K: [128][64] / V: [64][128]
  float* relk = (float*)(smem + SM_RELK);  // [9][64]
  float* relv = (float*)(smem + SM_RELV);  // [9][64]
  float* Ot   = (float*)(smem + SM_OT);    // [32][64]
  float* red  = (float*)(smem + SM_RED);   // [32][4]

  const bf16* qbh = qb + ((size_t)b * H_ + h) * T_ * HD_;   // (t,d)
  const bf16* kbh = kb + ((size_t)b * H_ + h) * T_ * HD_;   // (t,d)
  const bf16* vbh = vb + ((size_t)b * H_ + h) * HD_ * T_;   // (d,t)

  // Qt: 16B chunk per thread, 2 rounds (256 chunks total)
#pragma unroll
  for (int e = tid; e < 256; e += 128) {
    const int q = e >> 3, d0 = (e & 7) * 8;
    *(v8bf*)(Qt + q * 64 + d0) = *(const v8bf*)(qbh + (size_t)(q0 + q) * HD_ + d0);
  }
  for (int e = tid; e < 9 * 64; e += 128) {
    relk[e] = erk[(size_t)h * 9 * 64 + e];
    relv[e] = erv[(size_t)h * 9 * 64 + e];
  }
  __syncthreads();

  // Q fragments are loop-invariant: 2 q-subtiles x 2 d-chunks
  const v16bf qa00 = load_a_bf16(Qt, 64);
  const v16bf qa01 = load_a_bf16(Qt + 32, 64);
  const v16bf qa10 = load_a_bf16(Qt + 16 * 64, 64);
  const v16bf qa11 = load_a_bf16(Qt + 16 * 64 + 32, 64);
  const int mo = (lane >> 4) << 3;
  const int nn = lane & 15;

  // ---- pass 1: S = Qs * K^T  (8 WMMA per wave per 128-key slab) ----
  for (int k0 = 0; k0 < T_; k0 += 128) {
    __syncthreads();  // KVt reuse barrier
    {  // K tile [128 k][64 d]: one k-row per thread, 8x b128 in / 8x b128 out
      const bf16* gp = kbh + (size_t)(k0 + tid) * HD_;
      bf16* lp = KVt + tid * 64;
#pragma unroll
      for (int i = 0; i < 8; ++i) *(v8bf*)(lp + i * 8) = *(const v8bf*)(gp + i * 8);
    }
    __syncthreads();
#pragma unroll
    for (int f = 0; f < 2; ++f) {
      const int n0 = (wave * 2 + f) * 16;  // 8 k-subtiles over 4 waves
      const v16bf bk0 = load_bT_bf16(KVt + n0 * 64, 64);        // d 0..31
      const v16bf bk1 = load_bT_bf16(KVt + n0 * 64 + 32, 64);   // d 32..63
      {
        v8f acc = {};
        acc = wmma_bf(qa00, bk0, acc);
        acc = wmma_bf(qa01, bk1, acc);
#pragma unroll
        for (int r = 0; r < 8; ++r) S[(r + mo) * SLD_ + k0 + n0 + nn] = acc[r];
      }
      {
        v8f acc = {};
        acc = wmma_bf(qa10, bk0, acc);
        acc = wmma_bf(qa11, bk1, acc);
#pragma unroll
        for (int r = 0; r < 8; ++r) S[(16 + r + mo) * SLD_ + k0 + n0 + nn] = acc[r];
      }
    }
  }
  __syncthreads();

  // ---- rel-k band: k-q in [-3, +5], j = 3 + (k-q) ----
  for (int e = tid; e < 32 * 9; e += 128) {
    const int q = e / 9, j = e % 9;
    const int kg = q0 + q + (j - 3);
    if (kg >= 0 && kg < T_) {
      float dot = 0.f;
#pragma unroll
      for (int d = 0; d < 64; ++d) dot += (float)Qt[q * 64 + d] * relk[j * 64 + d];
      S[q * SLD_ + kg] += dot;   // Q already carries the 1/sqrt(HD) scale
    }
  }
  __syncthreads();

  // ---- mask + row softmax (4 threads per row, 256-col segments) ----
  const int row = tid >> 2, seg = tid & 3;
  const float* mrow = mask + ((size_t)b * T_ + (q0 + row)) * T_;
  float lmax = -3.0e38f;
  for (int k = seg * 256; k < seg * 256 + 256; ++k) {
    float v = S[row * SLD_ + k];
    if (mrow[k] == 0.f) { v = -1.0e4f; S[row * SLD_ + k] = v; }
    lmax = fmaxf(lmax, v);
  }
  red[row * 4 + seg] = lmax;
  __syncthreads();
  float rmax = red[row * 4];
#pragma unroll
  for (int i = 1; i < 4; ++i) rmax = fmaxf(rmax, red[row * 4 + i]);
  __syncthreads();
  float lsum = 0.f;
  for (int k = seg * 256; k < seg * 256 + 256; ++k) {
    const float ev = __expf(S[row * SLD_ + k] - rmax);
    S[row * SLD_ + k] = ev;
    lsum += ev;
  }
  red[row * 4 + seg] = lsum;
  __syncthreads();
  float rsum = 0.f;
#pragma unroll
  for (int i = 0; i < 4; ++i) rsum += red[row * 4 + i];
  const float inv = 1.0f / rsum;
  for (int k = seg * 256; k < seg * 256 + 256; ++k) S[row * SLD_ + k] *= inv;
  __syncthreads();

  // ---- pass 2: O = P * V  (8 WMMA per wave per slab, V-frag reused 2x) ----
  v8f oacc0 = {};  // q rows 0..15
  v8f oacc1 = {};  // q rows 16..31
  const int n0 = wave * 16;  // 4 d-subtiles over 4 waves
  for (int k0 = 0; k0 < T_; k0 += 128) {
    {  // V tile [64 d][128 k] from (d,t) global: contiguous both sides
      const int d = tid >> 1, half = (tid & 1) * 64;
      const bf16* gp = vbh + (size_t)d * T_ + k0 + half;
      bf16* lp = KVt + d * 128 + half;
#pragma unroll
      for (int i = 0; i < 8; ++i) *(v8bf*)(lp + i * 8) = *(const v8bf*)(gp + i * 8);
    }
    __syncthreads();
#pragma unroll
    for (int kc = 0; kc < 4; ++kc) {
      const v16bf vbf = load_bT_bf16(KVt + n0 * 128 + kc * 32, 128);
      const v16bf pa0 = load_a_f32(S + k0 + kc * 32, SLD_);
      const v16bf pa1 = load_a_f32(S + 16 * SLD_ + k0 + kc * 32, SLD_);
      oacc0 = wmma_bf(pa0, vbf, oacc0);
      oacc1 = wmma_bf(pa1, vbf, oacc1);
    }
    __syncthreads();
  }
#pragma unroll
  for (int r = 0; r < 8; ++r) {
    Ot[(r + mo) * 64 + n0 + nn] = oacc0[r];
    Ot[(16 + r + mo) * 64 + n0 + nn] = oacc1[r];
  }
  __syncthreads();

  // ---- rel-v band + writeout in (b, c, t) layout for the out-projection ----
  for (int e = tid; e < 32 * 64; e += 128) {
    const int q = e >> 6, d = e & 63;
    float val = Ot[q * 64 + d];
#pragma unroll
    for (int j = 0; j < 9; ++j) {
      const int kg = q0 + q + (j - 3);
      if (kg >= 0 && kg < T_) val += S[q * SLD_ + kg] * relv[j * 64 + d];
    }
    aout[((size_t)b * C_ + (h * 64 + d)) * T_ + (q0 + q)] = (bf16)val;
  }
}

// ---------------- Kernel 3: output projection (fp32 out) ----------------
// grid (T/128, OC/128, B), block 256 (8 waves), 128x128 tile
__global__ __launch_bounds__(256) void out_proj_kernel(
    const bf16* __restrict__ ain, const float* __restrict__ wo,
    const float* __restrict__ bo, float* __restrict__ out) {
  const int t0 = blockIdx.x * 128;
  const int o0 = blockIdx.y * 128;
  const int b = blockIdx.z;

  const int LD = 40;
  __shared__ bf16 Atile[128 * 40];  // [o][c]
  __shared__ bf16 Bt[128 * 40];     // [t][c]

  const int tid = threadIdx.x;
  const int wave = tid >> 5;
  const int msub = (wave & 3) * 32;
  const int nsub = (wave >> 2) * 64;

  v8f acc[2][4] = {};

  for (int kk = 0; kk < C_; kk += 32) {
#pragma unroll
    for (int e = tid; e < 512; e += 256) {
      const int r = e >> 2, ch = (e & 3) * 8;
      const float* gp = wo + (size_t)(o0 + r) * C_ + kk + ch;
      v8bf w;
#pragma unroll
      for (int i = 0; i < 8; ++i) w[i] = (bf16)gp[i];
      *(v8bf*)(Atile + r * LD + ch) = w;
    }
#pragma unroll
    for (int e = tid; e < 512; e += 256) {
      const int t = e & 127, ch = (e >> 7) * 8;
      const bf16* gp = ain + ((size_t)b * C_ + kk + ch) * T_ + t0 + t;
      v8bf w;
#pragma unroll
      for (int i = 0; i < 8; ++i) w[i] = gp[(size_t)i * T_];
      *(v8bf*)(Bt + t * LD + ch) = w;
    }
    __syncthreads();
    const v16bf a0 = load_a_bf16(Atile + msub * LD, LD);
    const v16bf a1 = load_a_bf16(Atile + (msub + 16) * LD, LD);
#pragma unroll
    for (int j = 0; j < 4; ++j) {
      const v16bf bj = load_bT_bf16(Bt + (nsub + j * 16) * LD, LD);
      acc[0][j] = wmma_bf(a0, bj, acc[0][j]);
      acc[1][j] = wmma_bf(a1, bj, acc[1][j]);
    }
    __syncthreads();
  }

  const int lane = tid & 31;
  const int mo = (lane >> 4) << 3;
  const int n = lane & 15;
#pragma unroll
  for (int i = 0; i < 2; ++i) {
#pragma unroll
    for (int j = 0; j < 4; ++j) {
#pragma unroll
      for (int r = 0; r < 8; ++r) {
        const int m = msub + i * 16 + r + mo;
        const int o = o0 + m;
        const int t = t0 + nsub + j * 16 + n;
        out[((size_t)b * OC_ + o) * T_ + t] = acc[i][j][r] + bo[o];
      }
    }
  }
}

// ---------------- Launch ----------------
extern "C" void kernel_launch(void* const* d_in, const int* in_sizes, int n_in,
                              void* d_out, int out_size, void* d_ws, size_t ws_size,
                              hipStream_t stream) {
  (void)in_sizes; (void)n_in; (void)out_size; (void)ws_size;
  const float* x    = (const float*)d_in[0];
  const float* c    = (const float*)d_in[1];
  const float* mask = (const float*)d_in[2];
  const float* wq   = (const float*)d_in[3];
  const float* bq   = (const float*)d_in[4];
  const float* wk   = (const float*)d_in[5];
  const float* bk   = (const float*)d_in[6];
  const float* wv   = (const float*)d_in[7];
  const float* bv   = (const float*)d_in[8];
  const float* wo   = (const float*)d_in[9];
  const float* bo   = (const float*)d_in[10];
  const float* erk  = (const float*)d_in[11];
  const float* erv  = (const float*)d_in[12];

  const size_t NQKV = (size_t)B_ * H_ * T_ * HD_;  // 3,145,728 elems
  bf16* qb = (bf16*)d_ws;
  bf16* kb = qb + NQKV;
  bf16* vb = kb + NQKV;
  bf16* ao = vb + NQKV;

  static bool attr_set = false;  // idempotent attribute config (not stream state)
  if (!attr_set) {
    hipFuncSetAttribute((const void*)attn_kernel,
                        hipFuncAttributeMaxDynamicSharedMemorySize, SM_TOTAL);
    attr_set = true;
  }

  dim3 g1(T_ / 128, C_ / 128, B_ * 3), b1(256);
  qkv_proj_kernel<<<g1, b1, 0, stream>>>(x, c, wq, bq, wk, bk, wv, bv, qb, kb, vb);

  dim3 g2(T_ / 32, H_, B_), b2(128);
  attn_kernel<<<g2, b2, SM_TOTAL, stream>>>(qb, kb, vb, mask, erk, erv, ao);

  dim3 g3(T_ / 128, OC_ / 128, B_), b3(256);
  out_proj_kernel<<<g3, b3, 0, stream>>>(ao, wo, bo, (float*)d_out);
}
